// CostVolumeManager_1726576856620
// MI455X (gfx1250) — matrix operand order repaired and checked by reading, not compiled, not run on hardware
//
#include <hip/hip_runtime.h>
#include <cmath>

#define H_ 96
#define W_ 128
#define D_ 64
#define B_ 8
#define K_ 7
#define C_ 32
#define HW_ (H_ * W_)
#define EPS_ 1e-8f

typedef __attribute__((ext_vector_type(16))) _Float16 v16h;
typedef __attribute__((ext_vector_type(8)))  float    v8f;
// 4-byte-aligned float pair: lets clang emit one global_load_b64 at dword alignment
typedef float f2u __attribute__((ext_vector_type(2), aligned(4)));

// ---------------------------------------------------------------------------
// Setup: per (b,k) fold projection into M(3x3), t(3); per (b,d) depth planes.
// ws layout (floats): [0 .. B*K*12) = {M row-major(9), t(3)} per (b,k)
//                     [B*K*12 .. B*K*12 + B*D) = depth planes
// ---------------------------------------------------------------------------
__global__ void cv_setup_kernel(const float* __restrict__ srcE,
                                const float* __restrict__ srcK,
                                const float* __restrict__ invK,
                                const float* __restrict__ mind,
                                const float* __restrict__ maxd,
                                float* __restrict__ ws) {
    int t = blockIdx.x * blockDim.x + threadIdx.x;
    if (t < B_ * K_) {
        int b = t / K_;
        const float* E  = srcE + t * 16;
        const float* Km = srcK + t * 16;
        const float* iK = invK + b * 16;
        float P[3][4];
        #pragma unroll
        for (int i = 0; i < 3; ++i)
            #pragma unroll
            for (int j = 0; j < 4; ++j) {
                float s = 0.f;
                #pragma unroll
                for (int l = 0; l < 4; ++l) s += Km[i * 4 + l] * E[l * 4 + j];
                P[i][j] = s;
            }
        float* o = ws + t * 12;
        #pragma unroll
        for (int i = 0; i < 3; ++i)
            #pragma unroll
            for (int j = 0; j < 3; ++j) {
                float s = 0.f;
                #pragma unroll
                for (int l = 0; l < 3; ++l) s += P[i][l] * iK[l * 4 + j];
                o[i * 3 + j] = s;
            }
        o[9]  = P[0][3];
        o[10] = P[1][3];
        o[11] = P[2][3];
    }
    int t2 = t - B_ * K_;
    if (t2 >= 0 && t2 < B_ * D_) {
        int b = t2 / D_;
        int d = t2 % D_;
        float mn = mind[b], mx = maxd[b];
        float ramp = (float)d / (float)(D_ - 1);
        ws[B_ * K_ * 12 + t2] = expf(logf(mn) + logf(mx / mn) * ramp);
    }
}

// ---------------------------------------------------------------------------
// Main kernel: one wave32 per 16-pixel row segment.
//   lane = h*16 + p :  p = pixel in group (0..15), h = channel-half (0/1)
// A (16x32 f16)  : warped src features, lane-half h holds K in {16h..16h+15}, row = p
// B (32x16 f16)  : cur features, lane-half hh holds K in {16hh..16hh+15}, col = p
// C (16x16 f32)  : accumulated over K views; diagonal = masked dot products
// ---------------------------------------------------------------------------
__global__ __launch_bounds__(256) void cost_volume_kernel(
    const float* __restrict__ cur,   // (B,C,H,W)
    const float* __restrict__ src,   // (B,K,C,H,W)
    const float* __restrict__ ws,
    float* __restrict__ out)         // (B,D,H,W)
{
    const int lane = threadIdx.x & 31;
    const int wid  = blockIdx.x * 8 + (threadIdx.x >> 5);
    const int GX   = W_ / 16;
    const int b    = wid / (H_ * GX);
    const int rem  = wid % (H_ * GX);
    const int y    = rem / GX;
    const int x0   = (rem % GX) * 16;
    const int h    = lane >> 4;
    const int p    = lane & 15;
    const int x    = x0 + p;

    // B matrix: current-frame features, channels 16h .. 16h+15 at (y,x).
    const float* curb = cur + (size_t)b * C_ * HW_ + y * W_ + x;
    v16h bvec;
    #pragma unroll
    for (int j = 0; j < 16; ++j)
        bvec[j] = (_Float16)curb[(size_t)(16 * h + j) * HW_];

    // Per-view projection of this pixel's ray: q = M * (px,py,1), t.
    const float px = (float)x + 0.5f;
    const float py = (float)y + 0.5f;
    float qx[K_], qy[K_], qz[K_], tx[K_], ty[K_], tz[K_];
    #pragma unroll
    for (int k = 0; k < K_; ++k) {
        const float* o = ws + (b * K_ + k) * 12;
        qx[k] = o[0] * px + o[1] * py + o[2];
        qy[k] = o[3] * px + o[4] * py + o[5];
        qz[k] = o[6] * px + o[7] * py + o[8];
        tx[k] = o[9]; ty[k] = o[10]; tz[k] = o[11];
    }

    const float* planes = ws + B_ * K_ * 12 + b * D_;
    const float* srcb   = src + (size_t)b * K_ * C_ * HW_;

    const bool owner = (h == 0) ? (p < 8) : (p >= 8);
    const int  idx   = p & 7;

    for (int d = 0; d < D_; ++d) {
        const float depth = planes[d];
        v8f acc = {};
        #pragma unroll
        for (int k = 0; k < K_; ++k) {
            const float cx = depth * qx[k] + tx[k];
            const float cy = depth * qy[k] + ty[k];
            const float cz = depth * qz[k] + tz[k];
            const float z  = cz + EPS_;
            const float sc = (fabsf(cz) > EPS_) ? (1.0f / z) : 1.0f;
            const float u  = cx * sc - 0.5f;   // == (gx+1)*0.5*W - 0.5
            const float v  = cy * sc - 0.5f;
            const float xf = floorf(u), yf = floorf(v);
            const float wx1 = u - xf, wy1 = v - yf;
            const float wx0 = 1.f - wx1, wy0 = 1.f - wy1;
            const int xi0 = (int)xf, yi0 = (int)yf;
            const int xi1 = xi0 + 1, yi1 = yi0 + 1;
            const float m   = (z > 0.f) ? 1.f : 0.f;
            const float vx0 = (xi0 >= 0 && xi0 <= W_ - 1) ? 1.f : 0.f;
            const float vx1 = (xi1 >= 0 && xi1 <= W_ - 1) ? 1.f : 0.f;
            const float vy0 = (yi0 >= 0 && yi0 <= H_ - 1) ? 1.f : 0.f;
            const float vy1 = (yi1 >= 0 && yi1 <= H_ - 1) ? 1.f : 0.f;

            // Fetch x-tap pairs with one b64 per row: base column xb, always
            // inside the row. Border shifts are folded into the pair weights.
            const int  xb     = min(max(xi0, 0), W_ - 2);
            const bool shiftL = (xi0 < 0);       // tap1 sits at pair[0]
            const bool shiftR = (xi0 > xb);      // tap0 sits at pair[1]
            const float wvx0 = wx0 * vx0;
            const float wvx1 = wx1 * vx1;
            const float a0 = shiftR ? 0.f  : (shiftL ? wvx1 : wvx0);
            const float a1 = shiftR ? wvx0 : (shiftL ? 0.f  : wvx1);
            const float wy0v = wy0 * vy0 * m;
            const float wy1v = wy1 * vy1 * m;
            const float w0x = a0 * wy0v, w0y = a1 * wy0v;
            const float w1x = a0 * wy1v, w1y = a1 * wy1v;

            const int yc0 = min(max(yi0, 0), H_ - 1);
            const int yc1 = min(max(yi1, 0), H_ - 1);
            const int o0  = yc0 * W_ + xb;
            const int o1  = yc1 * W_ + xb;
            const float* sk = srcb + (size_t)k * C_ * HW_;

            v16h avec;
            #pragma unroll
            for (int j = 0; j < 16; ++j) {
                // A-matrix channel for this lane: {16h..16h+15}
                const int c = 16 * h + j;
                const float* sc2 = sk + (size_t)c * HW_;
                const f2u pr0 = *(const f2u*)(sc2 + o0);
                const f2u pr1 = *(const f2u*)(sc2 + o1);
                const float val = w0x * pr0.x + w0y * pr0.y +
                                  w1x * pr1.x + w1y * pr1.y;
                avec[j] = (_Float16)val;
            }
            acc = __builtin_amdgcn_wmma_f32_16x16x32_f16(
                false, avec, false, bvec, (short)0, acc, false, false);
        }
        // Diagonal extract: p<8 -> (vgpr p, lane p); p>=8 -> (vgpr p-8, lane p+16)
        float r = acc[0];
        #pragma unroll
        for (int i = 1; i < 8; ++i)
            if (idx == i) r = acc[i];
        if (owner)
            out[(((size_t)b * D_ + d) * H_ + y) * W_ + x] = r;
    }
}

extern "C" void kernel_launch(void* const* d_in, const int* in_sizes, int n_in,
                              void* d_out, int out_size, void* d_ws, size_t ws_size,
                              hipStream_t stream) {
    const float* cur  = (const float*)d_in[0];
    const float* srcf = (const float*)d_in[1];
    const float* E    = (const float*)d_in[2];
    const float* Ks   = (const float*)d_in[3];
    const float* invK = (const float*)d_in[4];
    const float* mind = (const float*)d_in[5];
    const float* maxd = (const float*)d_in[6];
    float* out = (float*)d_out;
    float* ws  = (float*)d_ws;

    cv_setup_kernel<<<3, 256, 0, stream>>>(E, Ks, invK, mind, maxd, ws);

    const int groups = B_ * H_ * (W_ / 16);   // 6144 waves
    const int blocks = groups / 8;            // 8 waves per 256-thread block
    cost_volume_kernel<<<blocks, 256, 0, stream>>>(cur, srcf, ws, out);
}
